// MelDecoder_91018946937645
// MI455X (gfx1250) — compile-verified, matrix operand order checked by reading, not run on hardware
//
#include <hip/hip_runtime.h>
#include <hip/hip_bf16.h>
#include <math.h>

// ---------------------------------------------------------------------------
// MelDecoder (Tacotron attention decoder) for gfx1250 / MI455X.
//  - fp16 WMMA (v_wmma_f32_16x16x32_f16) for every GEMM, fp32 accumulate
//  - branch-free vectorized (b128) fragment loaders; ragged K pre-padded
//  - fp16 activation mirrors so WMMA inner loops have zero cvt / zero guard
//  - persistent decoder kernel (16 WGs x 256 thr), device generation barrier
//    + CDNA5 cluster-barrier fast path
//  - memory/keys fp16 -> 33MB attention working set, L2-resident (192MB L2)
// ---------------------------------------------------------------------------

typedef __attribute__((ext_vector_type(16))) _Float16 v16h;
typedef __attribute__((ext_vector_type(8)))  _Float16 v8h;
typedef __attribute__((ext_vector_type(8)))  float    v8f;

#define HDIM 256
#define PDIM 128
#define BATCH 64
#define MEMN 512
#define NMEL 80
#define KP1 96            // fc1 K padded 80 -> 96
#define RSTEP 5
#define TSTEPS 400
#define TFULL 2000
#define G3 768            // 3*HDIM
#define OUTC 400          // NMEL*RSTEP

// ------------------------- WMMA fragment loaders ---------------------------
// 16-bit A 16x32 layout (ISA 7.12.2): lanes 0-15 hold M=lane, K={0..7,16..23};
// lanes 16-31 hold K={8..15,24..31}. B (32x16) mirrors with lane = N.
// Each lane therefore needs two contiguous 8-element runs -> 2 x b128 (fp16)
// or 4 x b128 (fp32 source).

__device__ __forceinline__ v16h load_fragh(const _Float16* P, int ld, int rc0, int k0) {
    int lane = threadIdx.x & 31;
    const _Float16* base = P + (size_t)(rc0 + (lane & 15)) * ld
                             + k0 + ((lane & 16) ? 8 : 0);
    v8h lo = *(const v8h*)(base);        // K run {0..7}
    v8h hi = *(const v8h*)(base + 16);   // K run {16..23}
    v16h f;
#pragma unroll
    for (int i = 0; i < 8; ++i) { f[i] = lo[i]; f[i + 8] = hi[i]; }
    return f;
}

__device__ __forceinline__ v16h load_fragf(const float* A, int lda, int row0, int k0) {
    int lane = threadIdx.x & 31;
    const float* base = A + (size_t)(row0 + (lane & 15)) * lda
                          + k0 + ((lane & 16) ? 8 : 0);
    float4 a0 = *(const float4*)(base);
    float4 a1 = *(const float4*)(base + 4);
    float4 b0 = *(const float4*)(base + 16);
    float4 b1 = *(const float4*)(base + 20);
    v16h f;
    f[0] = (_Float16)a0.x; f[1] = (_Float16)a0.y; f[2]  = (_Float16)a0.z; f[3]  = (_Float16)a0.w;
    f[4] = (_Float16)a1.x; f[5] = (_Float16)a1.y; f[6]  = (_Float16)a1.z; f[7]  = (_Float16)a1.w;
    f[8] = (_Float16)b0.x; f[9] = (_Float16)b0.y; f[10] = (_Float16)b0.z; f[11] = (_Float16)b0.w;
    f[12] = (_Float16)b1.x; f[13] = (_Float16)b1.y; f[14] = (_Float16)b1.z; f[15] = (_Float16)b1.w;
    return f;
}

// ----------------------------- prep kernels --------------------------------

__global__ void k_f2h(const float* __restrict__ s, _Float16* __restrict__ d, int n) {
    int i = blockIdx.x * blockDim.x + threadIdx.x;
    if (i < n) d[i] = (_Float16)s[i];
}

// convert (rows x kin) fp32 -> (rows x kout) fp16, zero-padded in K
__global__ void k_f2h_pad(const float* __restrict__ s, _Float16* __restrict__ d,
                          int rows, int kin, int kout) {
    int idx = blockIdx.x * blockDim.x + threadIdx.x;
    if (idx >= rows * kout) return;
    int r = idx / kout, k = idx - r * kout;
    d[idx] = (k < kin) ? (_Float16)s[(size_t)r * kin + k] : (_Float16)0.0f;
}

__global__ void k_init(float* states, _Float16* states16, unsigned* sync) {
    int i = blockIdx.x * blockDim.x + threadIdx.x;
    if (i < 3 * BATCH * HDIM) { states[i] = 0.0f; states16[i] = (_Float16)0.0f; }
    if (i == 0) { sync[0] = 0u; sync[1] = 0u; }
}

// frames[(t*64+b)*96 + k] = din[b,k,t*R], zero pad k in [80,96)
__global__ void k_gather_frames(const float* __restrict__ din, float* __restrict__ fr) {
    int idx = blockIdx.x * blockDim.x + threadIdx.x;
    if (idx >= TSTEPS * BATCH * KP1) return;
    int row = idx / KP1, k = idx - row * KP1;
    int t = row >> 6, b = row & 63;
    fr[idx] = (k < NMEL)
        ? din[(size_t)b * NMEL * TFULL + (size_t)k * TFULL + (size_t)t * RSTEP]
        : 0.0f;
}

// Tiled WMMA GEMM: OUT[m,n] = act( A(MxK fp32) . W(NxK fp16)^T + bias )
__global__ void k_gemm(const float* __restrict__ A, int lda,
                       const _Float16* __restrict__ W, int ldw,
                       const float* __restrict__ bias,
                       float* __restrict__ outF, _Float16* __restrict__ outH,
                       int ldo, int Mrows, int N, int K, int relu) {
    int lane = threadIdx.x & 31;
    int wid  = (blockIdx.x * blockDim.x + threadIdx.x) >> 5;
    int mt = Mrows >> 4;
    int ntiles = mt * (N >> 4);
    if (wid >= ntiles) return;
    int row0 = (wid % mt) << 4, col0 = (wid / mt) << 4;
    v8f acc = {};
    for (int k0 = 0; k0 < K; k0 += 32) {
        v16h a = load_fragf(A, lda, row0, k0);
        v16h b = load_fragh(W, ldw, col0, k0);
        acc = __builtin_amdgcn_wmma_f32_16x16x32_f16(false, a, false, b,
                                                     (short)0, acc, false, false);
    }
    int n  = col0 + (lane & 15);
    int m0 = row0 + ((lane & 16) ? 8 : 0);
    float bv = bias ? bias[n] : 0.0f;
#pragma unroll
    for (int r = 0; r < 8; ++r) {
        float v = acc[r] + bv;
        if (relu) v = fmaxf(v, 0.0f);
        if (outF) outF[(size_t)(m0 + r) * ldo + n] = v;
        else      outH[(size_t)(m0 + r) * ldo + n] = (_Float16)v;
    }
}

// ------------------------- persistent decoder ------------------------------

struct DecP {
    float* out;                  // (B,80,2000)
    const _Float16* x_tf_h;      // (T,B,P) fp16
    const _Float16* keys_h;      // (B,M,H)
    const _Float16* mem_h;       // (B,M,H)
    const float* v_w;            // (H)
    const _Float16 *a_wih_h, *a_whh_h, *g1_wih_h, *g1_whh_h,
                   *g2_wih_h, *g2_whh_h, *W2_h, *proj_h, *outw_h;
    const float *a_bih, *a_bhh, *g1_bih, *g1_bhh, *g2_bih, *g2_bhh,
                *proj_b, *out_b;
    // fp32 state (recurrence precision) + fp16 mirrors (WMMA A operands)
    float *h_attn, *h_g1, *h_g2;
    _Float16 *h_attn16, *h_g116, *h_g216;
    float *g1_in, *g2_in;
    _Float16 *g1_in16, *g2_in16, *ctx16, *bf16;
    float *gi, *gh, *query, *scores, *attnw;
    unsigned* sync;              // [0]=counter [1]=generation
    int nwg;
};

__device__ __forceinline__ float sigmoidf_(float x) { return 1.0f / (1.0f + expf(-x)); }

__device__ void grid_sync(const DecP& p) {
    __threadfence();
    __syncthreads();
    // CDNA5 cluster barrier: HW NOP when ClusterID==0, cheap sync if clustered.
    __builtin_amdgcn_s_cluster_barrier();
    if (threadIdx.x == 0) {
        unsigned g = __atomic_load_n(&p.sync[1], __ATOMIC_ACQUIRE);
        unsigned v = __atomic_add_fetch(&p.sync[0], 1u, __ATOMIC_ACQ_REL);
        if (v == (unsigned)p.nwg) {
            __atomic_store_n(&p.sync[0], 0u, __ATOMIC_RELAXED);
            __atomic_store_n(&p.sync[1], g + 1u, __ATOMIC_RELEASE);
        } else {
            while (__atomic_load_n(&p.sync[1], __ATOMIC_ACQUIRE) == g)
                __builtin_amdgcn_s_sleep(1);
        }
    }
    __syncthreads();
}

// OUT(64 x N) = A(64 x K fp16) . W(N x K fp16)^T + bias (wave-uniform tiles)
__device__ void gemm_tiles(const _Float16* A, const _Float16* W,
                           const float* bias, float* OUT, int N, int K,
                           int wid, int nw, int lane) {
    int ntiles = 4 * (N >> 4);
    for (int tj = wid; tj < ntiles; tj += nw) {
        int row0 = (tj & 3) << 4, col0 = (tj >> 2) << 4;
        v8f acc = {};
        for (int k0 = 0; k0 < K; k0 += 32) {
            v16h a = load_fragh(A, K, row0, k0);
            v16h b = load_fragh(W, K, col0, k0);
            acc = __builtin_amdgcn_wmma_f32_16x16x32_f16(false, a, false, b,
                                                         (short)0, acc, false, false);
        }
        int n  = col0 + (lane & 15);
        int m0 = row0 + ((lane & 16) ? 8 : 0);
        float bv = bias ? bias[n] : 0.0f;
#pragma unroll
        for (int r = 0; r < 8; ++r)
            OUT[(size_t)(m0 + r) * N + n] = acc[r] + bv;
    }
}

// g1_in(64x256) = [A1|A2](64x512 fp16) . proj(256x512)^T + bias, dual output
__device__ void gemm_concat(const _Float16* A1, const _Float16* A2,
                            const _Float16* W, const float* bias,
                            float* OUT, _Float16* OUTH,
                            int wid, int nw, int lane) {
    for (int tj = wid; tj < 64; tj += nw) {
        int row0 = (tj & 3) << 4, col0 = (tj >> 2) << 4;
        v8f acc = {};
        for (int k0 = 0; k0 < 2 * HDIM; k0 += 32) {
            const _Float16* src = (k0 < HDIM) ? A1 : A2;
            int kk = (k0 < HDIM) ? k0 : k0 - HDIM;
            v16h a = load_fragh(src, HDIM, row0, kk);
            v16h b = load_fragh(W, 2 * HDIM, col0, k0);
            acc = __builtin_amdgcn_wmma_f32_16x16x32_f16(false, a, false, b,
                                                         (short)0, acc, false, false);
        }
        int n  = col0 + (lane & 15);
        int m0 = row0 + ((lane & 16) ? 8 : 0);
        float bv = bias[n];
#pragma unroll
        for (int r = 0; r < 8; ++r) {
            float v = acc[r] + bv;
            OUT[(size_t)(m0 + r) * HDIM + n]  = v;
            OUTH[(size_t)(m0 + r) * HDIM + n] = (_Float16)v;
        }
    }
}

// GRU gate combine: h updated in place (fp32) + fp16 mirror; optional residual
__device__ void gru_gate(const float* gi, const float* gh,
                         float* h, _Float16* h16,
                         const float* rin, float* rout, _Float16* rout16,
                         int gtid, int gsz) {
    for (int idx = gtid; idx < BATCH * HDIM; idx += gsz) {
        int b = idx >> 8, c = idx & 255;
        const float* gib = gi + (size_t)b * G3;
        const float* ghb = gh + (size_t)b * G3;
        float r = sigmoidf_(gib[c] + ghb[c]);
        float z = sigmoidf_(gib[HDIM + c] + ghb[HDIM + c]);
        float n = tanhf(gib[2 * HDIM + c] + r * ghb[2 * HDIM + c]);
        float hn = (1.0f - z) * n + z * h[idx];
        h[idx] = hn;
        h16[idx] = (_Float16)hn;
        if (rout16) {
            float rv = rin[idx] + hn;
            rout16[idx] = (_Float16)rv;
            if (rout) rout[idx] = rv;
        }
    }
}

__global__ void __launch_bounds__(256)
k_decoder(DecP p) {
    const int lane = threadIdx.x & 31;
    const int wid  = (blockIdx.x * blockDim.x + threadIdx.x) >> 5;
    const int nw   = (gridDim.x * blockDim.x) >> 5;
    const int gtid = blockIdx.x * blockDim.x + threadIdx.x;
    const int gsz  = gridDim.x * blockDim.x;

    for (int t = 0; t < TSTEPS; ++t) {
        const _Float16* xt = p.x_tf_h + (size_t)t * BATCH * PDIM;

        // ---- attention GRU pre-gates ----------------------------------
        gemm_tiles(xt, p.a_wih_h, p.a_bih, p.gi, G3, PDIM, wid, nw, lane);
        gemm_tiles(p.h_attn16, p.a_whh_h, p.a_bhh, p.gh, G3, HDIM, wid, nw, lane);
        grid_sync(p);
        gru_gate(p.gi, p.gh, p.h_attn, p.h_attn16, nullptr, nullptr, nullptr,
                 gtid, gsz);                       // h_attn = d_t
        grid_sync(p);

        // ---- query = d_t @ W2^T ---------------------------------------
        gemm_tiles(p.h_attn16, p.W2_h, nullptr, p.query, HDIM, HDIM, wid, nw, lane);
        grid_sync(p);

        // ---- scores[b,m] = tanh(keys + q) . v  (wave per row) ---------
        for (int row = wid; row < BATCH * MEMN; row += nw) {
            int b = row >> 9;
            v8h kv = *(const v8h*)(p.keys_h + (size_t)row * HDIM + lane * 8);
            const float* qb = p.query + (size_t)b * HDIM + lane * 8;
            const float* vb = p.v_w + lane * 8;
            float4 q0 = *(const float4*)(qb);
            float4 q1 = *(const float4*)(qb + 4);
            float4 v0 = *(const float4*)(vb);
            float4 v1 = *(const float4*)(vb + 4);
            float s = tanhf((float)kv[0] + q0.x) * v0.x
                    + tanhf((float)kv[1] + q0.y) * v0.y
                    + tanhf((float)kv[2] + q0.z) * v0.z
                    + tanhf((float)kv[3] + q0.w) * v0.w
                    + tanhf((float)kv[4] + q1.x) * v1.x
                    + tanhf((float)kv[5] + q1.y) * v1.y
                    + tanhf((float)kv[6] + q1.z) * v1.z
                    + tanhf((float)kv[7] + q1.w) * v1.w;
#pragma unroll
            for (int off = 16; off > 0; off >>= 1) s += __shfl_xor(s, off, 32);
            if (lane == 0) p.scores[row] = s;
        }
        grid_sync(p);

        // ---- softmax over M (wave per batch row) ----------------------
        for (int b = wid; b < BATCH; b += nw) {
            const float* sr = p.scores + (size_t)b * MEMN;
            float mx = -3.4e38f;
            for (int m = lane; m < MEMN; m += 32) mx = fmaxf(mx, sr[m]);
#pragma unroll
            for (int off = 16; off > 0; off >>= 1) mx = fmaxf(mx, __shfl_xor(mx, off, 32));
            float sum = 0.0f;
            for (int m = lane; m < MEMN; m += 32) {
                float e = expf(sr[m] - mx);
                p.attnw[(size_t)b * MEMN + m] = e;
                sum += e;
            }
#pragma unroll
            for (int off = 16; off > 0; off >>= 1) sum += __shfl_xor(sum, off, 32);
            float inv = 1.0f / sum;
            for (int m = lane; m < MEMN; m += 32)
                p.attnw[(size_t)b * MEMN + m] *= inv;
        }
        grid_sync(p);

        // ---- context = attnw @ memory -> fp16 (job = (b, 32-h block)) -
        for (int job = wid; job < BATCH * 8; job += nw) {
            int b = job >> 3, h = ((job & 7) << 5) + lane;
            const _Float16* mb = p.mem_h + (size_t)b * MEMN * HDIM + h;
            const float* wv = p.attnw + (size_t)b * MEMN;
            float acc = 0.0f;
            for (int m = 0; m < MEMN; ++m)
                acc += wv[m] * (float)mb[(size_t)m * HDIM];
            p.ctx16[(size_t)b * HDIM + h] = (_Float16)acc;
        }
        grid_sync(p);

        // ---- g1_in = [d_t | ctx] @ proj^T + b -------------------------
        gemm_concat(p.h_attn16, p.ctx16, p.proj_h, p.proj_b,
                    p.g1_in, p.g1_in16, wid, nw, lane);
        grid_sync(p);

        // ---- GRU1 -----------------------------------------------------
        gemm_tiles(p.g1_in16, p.g1_wih_h, p.g1_bih, p.gi, G3, HDIM, wid, nw, lane);
        gemm_tiles(p.h_g116, p.g1_whh_h, p.g1_bhh, p.gh, G3, HDIM, wid, nw, lane);
        grid_sync(p);
        gru_gate(p.gi, p.gh, p.h_g1, p.h_g116, p.g1_in, p.g2_in, p.g2_in16,
                 gtid, gsz);                       // g2_in = g1_in + h_g1
        grid_sync(p);

        // ---- GRU2 -----------------------------------------------------
        gemm_tiles(p.g2_in16, p.g2_wih_h, p.g2_bih, p.gi, G3, HDIM, wid, nw, lane);
        gemm_tiles(p.h_g216, p.g2_whh_h, p.g2_bhh, p.gh, G3, HDIM, wid, nw, lane);
        grid_sync(p);
        gru_gate(p.gi, p.gh, p.h_g2, p.h_g216, p.g2_in, nullptr, p.bf16,
                 gtid, gsz);                       // bf = g2_in + h_g2
        grid_sync(p);

        // ---- out = bf @ out_w^T + b, scattered to (B,80,2000) ---------
        for (int tj = wid; tj < 100; tj += nw) {
            int row0 = (tj & 3) << 4, col0 = (tj >> 2) << 4;
            v8f acc = {};
            for (int k0 = 0; k0 < HDIM; k0 += 32) {
                v16h a = load_fragh(p.bf16, HDIM, row0, k0);
                v16h b = load_fragh(p.outw_h, HDIM, col0, k0);
                acc = __builtin_amdgcn_wmma_f32_16x16x32_f16(false, a, false, b,
                                                             (short)0, acc, false, false);
            }
            int o  = col0 + (lane & 15);
            int m0 = row0 + ((lane & 16) ? 8 : 0);
            float bv = p.out_b[o];
            int mel = o / RSTEP, rr = o - RSTEP * mel;
#pragma unroll
            for (int r = 0; r < 8; ++r) {
                int b = m0 + r;
                p.out[(size_t)b * (NMEL * TFULL) + (size_t)mel * TFULL
                      + (size_t)t * RSTEP + rr] = acc[r] + bv;
            }
        }
        grid_sync(p);
    }
}

// ------------------------------ host side ----------------------------------

extern "C" void kernel_launch(void* const* d_in, const int* in_sizes, int n_in,
                              void* d_out, int out_size, void* d_ws, size_t ws_size,
                              hipStream_t stream) {
    const float* din    = (const float*)d_in[0];
    const float* memory = (const float*)d_in[1];
    const float* fc1_w  = (const float*)d_in[2];
    const float* fc1_b  = (const float*)d_in[3];
    const float* fc2_w  = (const float*)d_in[4];
    const float* fc2_b  = (const float*)d_in[5];
    const float* W1_w   = (const float*)d_in[6];
    const float* W2_w   = (const float*)d_in[7];
    const float* v_w    = (const float*)d_in[8];
    const float* a_wih  = (const float*)d_in[9];
    const float* a_whh  = (const float*)d_in[10];
    const float* a_bih  = (const float*)d_in[11];
    const float* a_bhh  = (const float*)d_in[12];
    const float* g1_wih = (const float*)d_in[13];
    const float* g1_whh = (const float*)d_in[14];
    const float* g1_bih = (const float*)d_in[15];
    const float* g1_bhh = (const float*)d_in[16];
    const float* g2_wih = (const float*)d_in[17];
    const float* g2_whh = (const float*)d_in[18];
    const float* g2_bih = (const float*)d_in[19];
    const float* g2_bhh = (const float*)d_in[20];
    const float* proj_w = (const float*)d_in[21];
    const float* proj_b = (const float*)d_in[22];
    const float* out_w  = (const float*)d_in[23];
    const float* out_b  = (const float*)d_in[24];

    char* wp = (char*)d_ws;
    auto take = [&](size_t bytes) {
        char* p = wp;
        wp += (bytes + 255) & ~(size_t)255;
        return p;
    };

    unsigned* sync  = (unsigned*)take(256);
    float* states   = (float*)take((size_t)3 * BATCH * HDIM * 4);
    _Float16* st16  = (_Float16*)take((size_t)3 * BATCH * HDIM * 2);
    float* h_attn = states, *h_g1 = states + BATCH * HDIM, *h_g2 = states + 2 * BATCH * HDIM;
    _Float16* h_attn16 = st16, *h_g116 = st16 + BATCH * HDIM, *h_g216 = st16 + 2 * BATCH * HDIM;

    float*    g1_in   = (float*)take((size_t)BATCH * HDIM * 4);
    float*    g2_in   = (float*)take((size_t)BATCH * HDIM * 4);
    _Float16* g1_in16 = (_Float16*)take((size_t)BATCH * HDIM * 2);
    _Float16* g2_in16 = (_Float16*)take((size_t)BATCH * HDIM * 2);
    _Float16* ctx16   = (_Float16*)take((size_t)BATCH * HDIM * 2);
    _Float16* bf16    = (_Float16*)take((size_t)BATCH * HDIM * 2);
    float* gi     = (float*)take((size_t)BATCH * G3 * 4);
    float* gh     = (float*)take((size_t)BATCH * G3 * 4);
    float* query  = (float*)take((size_t)BATCH * HDIM * 4);
    float* scores = (float*)take((size_t)BATCH * MEMN * 4);
    float* attnw  = (float*)take((size_t)BATCH * MEMN * 4);

    _Float16* a_wih_h  = (_Float16*)take((size_t)G3 * PDIM * 2);
    _Float16* a_whh_h  = (_Float16*)take((size_t)G3 * HDIM * 2);
    _Float16* g1_wih_h = (_Float16*)take((size_t)G3 * HDIM * 2);
    _Float16* g1_whh_h = (_Float16*)take((size_t)G3 * HDIM * 2);
    _Float16* g2_wih_h = (_Float16*)take((size_t)G3 * HDIM * 2);
    _Float16* g2_whh_h = (_Float16*)take((size_t)G3 * HDIM * 2);
    _Float16* W2_h     = (_Float16*)take((size_t)HDIM * HDIM * 2);
    _Float16* proj_h   = (_Float16*)take((size_t)HDIM * 2 * HDIM * 2);
    _Float16* outw_h   = (_Float16*)take((size_t)OUTC * HDIM * 2);
    _Float16* fc1p_h   = (_Float16*)take((size_t)HDIM * KP1 * 2);   // padded 256x96
    _Float16* fc2_h    = (_Float16*)take((size_t)PDIM * HDIM * 2);
    _Float16* W1_h     = (_Float16*)take((size_t)HDIM * HDIM * 2);
    _Float16* mem_h    = (_Float16*)take((size_t)BATCH * MEMN * HDIM * 2);
    _Float16* keys_h   = (_Float16*)take((size_t)BATCH * MEMN * HDIM * 2);

    float*    frames = (float*)take((size_t)TSTEPS * BATCH * KP1 * 4);   // padded stride 96
    float*    h1buf  = (float*)take((size_t)TSTEPS * BATCH * HDIM * 4);
    _Float16* x_tf_h = (_Float16*)take((size_t)TSTEPS * BATCH * PDIM * 2);

    // --- init state + barrier ---
    k_init<<<(3 * BATCH * HDIM + 255) / 256, 256, 0, stream>>>(states, st16, sync);

    // --- weight / memory fp16 conversion ---
    auto cvt = [&](const float* s, _Float16* d, int n) {
        k_f2h<<<(n + 255) / 256, 256, 0, stream>>>(s, d, n);
    };
    cvt(a_wih,  a_wih_h,  G3 * PDIM);
    cvt(a_whh,  a_whh_h,  G3 * HDIM);
    cvt(g1_wih, g1_wih_h, G3 * HDIM);
    cvt(g1_whh, g1_whh_h, G3 * HDIM);
    cvt(g2_wih, g2_wih_h, G3 * HDIM);
    cvt(g2_whh, g2_whh_h, G3 * HDIM);
    cvt(W2_w,   W2_h,     HDIM * HDIM);
    cvt(proj_w, proj_h,   HDIM * 2 * HDIM);
    cvt(out_w,  outw_h,   OUTC * HDIM);
    cvt(fc2_w,  fc2_h,    PDIM * HDIM);
    cvt(W1_w,   W1_h,     HDIM * HDIM);
    cvt(memory, mem_h,    BATCH * MEMN * HDIM);
    {   // fc1 zero-padded to K=96
        int n = HDIM * KP1;
        k_f2h_pad<<<(n + 255) / 256, 256, 0, stream>>>(fc1_w, fc1p_h, HDIM, NMEL, KP1);
    }

    // --- prenet (only every R-th frame is consumed) ---
    {
        int n = TSTEPS * BATCH * KP1;
        k_gather_frames<<<(n + 255) / 256, 256, 0, stream>>>(din, frames);
    }
    {   // h1 = relu(frames @ fc1^T + b1): M=25600, N=256, K=96 (padded)
        int ntiles = (TSTEPS * BATCH / 16) * (HDIM / 16);
        k_gemm<<<(ntiles + 7) / 8, 256, 0, stream>>>(
            frames, KP1, fc1p_h, KP1, fc1_b, h1buf, nullptr, HDIM,
            TSTEPS * BATCH, HDIM, KP1, 1);
    }
    {   // x_tf = relu(h1 @ fc2^T + b2) -> fp16: M=25600, N=128, K=256
        int ntiles = (TSTEPS * BATCH / 16) * (PDIM / 16);
        k_gemm<<<(ntiles + 7) / 8, 256, 0, stream>>>(
            h1buf, HDIM, fc2_h, HDIM, fc2_b, nullptr, x_tf_h, PDIM,
            TSTEPS * BATCH, PDIM, HDIM, 1);
    }
    {   // keys = memory @ W1^T -> fp16: M=32768, N=256, K=256
        int ntiles = (BATCH * MEMN / 16) * (HDIM / 16);
        k_gemm<<<(ntiles + 7) / 8, 256, 0, stream>>>(
            memory, HDIM, W1_h, HDIM, nullptr, nullptr, keys_h, HDIM,
            BATCH * MEMN, HDIM, HDIM, 0);
    }

    // --- persistent sequential decoder ---
    DecP p;
    p.out = (float*)d_out;
    p.x_tf_h = x_tf_h; p.keys_h = keys_h; p.mem_h = mem_h; p.v_w = v_w;
    p.a_wih_h = a_wih_h; p.a_whh_h = a_whh_h;
    p.g1_wih_h = g1_wih_h; p.g1_whh_h = g1_whh_h;
    p.g2_wih_h = g2_wih_h; p.g2_whh_h = g2_whh_h;
    p.W2_h = W2_h; p.proj_h = proj_h; p.outw_h = outw_h;
    p.a_bih = a_bih; p.a_bhh = a_bhh;
    p.g1_bih = g1_bih; p.g1_bhh = g1_bhh;
    p.g2_bih = g2_bih; p.g2_bhh = g2_bhh;
    p.proj_b = proj_b; p.out_b = out_b;
    p.h_attn = h_attn; p.h_g1 = h_g1; p.h_g2 = h_g2;
    p.h_attn16 = h_attn16; p.h_g116 = h_g116; p.h_g216 = h_g216;
    p.g1_in = g1_in; p.g2_in = g2_in;
    p.g1_in16 = g1_in16; p.g2_in16 = g2_in16; p.ctx16 = ctx16; p.bf16 = bf16;
    p.gi = gi; p.gh = gh; p.query = query; p.scores = scores; p.attnw = attnw;
    p.sync = sync;
    p.nwg = 16;

    k_decoder<<<16, 256, 0, stream>>>(p);
}
